// TransformerLayer_79224966742239
// MI455X (gfx1250) — matrix operand order repaired and checked by reading, not compiled, not run on hardware
//
#include <hip/hip_runtime.h>
#include <cstddef>
#include <cstdint>

// ---------------------------------------------------------------------------
// Problem constants (fixed by the reference)
// ---------------------------------------------------------------------------
#define BATCH 8
#define SEQ   1024
#define DMODEL 1024
#define NHEAD 16
#define DHEAD 64
#define DFF   4096
#define MROWS (BATCH * SEQ)   // 8192

typedef __attribute__((ext_vector_type(16))) __bf16 v16bf;
typedef __attribute__((ext_vector_type(8)))  float  v8f;

// ---------------------------------------------------------------------------
// Scalar bf16 <-> f32 helpers (bit ops only; buffers are unsigned short)
// ---------------------------------------------------------------------------
__device__ __forceinline__ unsigned short f2bf(float f) {
    union { float f; unsigned u; } x; x.f = f;
    unsigned r = x.u + 0x7FFFu + ((x.u >> 16) & 1u);
    return (unsigned short)(r >> 16);
}

__device__ __forceinline__ v8f zero8() {
    v8f z;
#pragma unroll
    for (int i = 0; i < 8; ++i) z[i] = 0.0f;
    return z;
}

// ---------------------------------------------------------------------------
// gfx1250 async global->LDS copy (ASYNCcnt path), per CDNA5 ISA
// VDST = LDS byte address (low 32 bits of the generic shared pointer),
// VADDR = 64-bit global address, GV addressing mode.
// ---------------------------------------------------------------------------
__device__ __forceinline__ void async_load_b128(unsigned int lds_off, const void* gptr) {
    asm volatile("global_load_async_to_lds_b128 %0, %1, off"
                 :: "v"(lds_off), "v"((unsigned long long)(size_t)gptr)
                 : "memory");
}
__device__ __forceinline__ void wait_async0() {
    asm volatile("s_wait_asynccnt 0x0" ::: "memory");
}

// ---------------------------------------------------------------------------
// WMMA fragment loaders (contiguous k order: k = 16*(lane/16) + e, wave32)
// ---------------------------------------------------------------------------
// A-matrix 16x32 bf16, row-major source [*, lda]:
//   lanes 0-15:  M=lane,    elems 0-15 -> K=k0..k0+15 contiguous, k0=kbase
//   lanes 16-31: M=lane-16, k0 = kbase+16
__device__ __forceinline__ v16bf load_a_frag(const unsigned short* A, int lda,
                                             int mbase, int kbase, int lane) {
    const unsigned short* p =
        A + (size_t)(mbase + (lane & 15)) * lda + kbase + ((lane & 16) ? 16 : 0);
    union { v16bf v; uint4 q[2]; } u;
    u.q[0] = *reinterpret_cast<const uint4*>(p);
    u.q[1] = *reinterpret_cast<const uint4*>(p + 8);
    return u.v;
}

// B-matrix 32x16 bf16 loaded from a ROW-MAJOR TRANSPOSED source BT[N, Kdim]:
//   lane n = nbase + (lane&15); K = kbase + (lane<16 ? 0 : 16) + e, e contiguous.
__device__ __forceinline__ v16bf load_bt_frag(const unsigned short* BT, int ldbt,
                                              int nbase, int kbase, int lane) {
    const unsigned short* p =
        BT + (size_t)(nbase + (lane & 15)) * ldbt + kbase + ((lane & 16) ? 16 : 0);
    union { v16bf v; uint4 q[2]; } u;
    u.q[0] = *reinterpret_cast<const uint4*>(p);
    u.q[1] = *reinterpret_cast<const uint4*>(p + 8);
    return u.v;
}

// B-matrix fragment from pre-packed weights: 512 bf16 per (kt, nt) tile,
// laid out so lane L reads 16 contiguous bf16 (32 bytes).
__device__ __forceinline__ v16bf load_bp_frag(const unsigned short* Bp, int ntiles16,
                                              int kt, int nt, int lane) {
    const unsigned short* p = Bp + ((((size_t)kt * ntiles16 + nt) * 32 + lane) << 4);
    union { v16bf v; uint4 q[2]; } u;
    u.q[0] = *reinterpret_cast<const uint4*>(p);
    u.q[1] = *reinterpret_cast<const uint4*>(p + 8);
    return u.v;
}

__device__ __forceinline__ v8f wmma_bf16(v16bf a, v16bf b, v8f c) {
    // (neg_a, A, neg_b, B, c_mod, C, reuse_a, reuse_b)
    return __builtin_amdgcn_wmma_f32_16x16x32_bf16(false, a, false, b, (short)0, c,
                                                   false, false);
}

// ---------------------------------------------------------------------------
// Elementwise convert f32 -> bf16
// ---------------------------------------------------------------------------
__global__ void f32_to_bf16_kernel(const float* __restrict__ in,
                                   unsigned short* __restrict__ out, int n) {
    int i = blockIdx.x * blockDim.x + threadIdx.x;
    if (i < n) out[i] = f2bf(in[i]);
}

// ---------------------------------------------------------------------------
// Weight packing: row-major [K, N] f32 -> WMMA B-fragment-ordered bf16
// dst[((kt*(N/16)+nt)*32 + lane)*16 + e] = src[(kt*32 + klo(lane) + e) * N + nt*16 + lane%16]
// ---------------------------------------------------------------------------
__global__ void pack_b_std_kernel(const float* __restrict__ src,
                                  unsigned short* __restrict__ dst, int K, int N) {
    int idx = blockIdx.x * blockDim.x + threadIdx.x;
    if (idx >= K * N) return;
    int e    = idx & 15;
    int lane = (idx >> 4) & 31;
    int t    = idx >> 9;
    int ntiles = N >> 4;
    int nt = t % ntiles;
    int kt = t / ntiles;
    int n = (nt << 4) + (lane & 15);
    int k = (kt << 5) + ((lane >> 4) << 4) + e;
    dst[idx] = f2bf(src[(size_t)k * N + n]);
}

// QKV weights come in as [H, D, DK] (h, d, k). Flattened column n = h*DK + dk.
__global__ void pack_b_qkv_kernel(const float* __restrict__ src,
                                  unsigned short* __restrict__ dst) {
    // K = DMODEL = 1024, N = NHEAD*DHEAD = 1024
    int idx = blockIdx.x * blockDim.x + threadIdx.x;
    if (idx >= DMODEL * DMODEL) return;
    int e    = idx & 15;
    int lane = (idx >> 4) & 31;
    int t    = idx >> 9;
    int nt = t & 63;          // N/16 = 64
    int kt = t >> 6;
    int n = (nt << 4) + (lane & 15);
    int k = (kt << 5) + ((lane >> 4) << 4) + e;
    int h  = n >> 6;
    int dk = n & 63;
    dst[idx] = f2bf(src[(size_t)h * DMODEL * DHEAD + (size_t)k * DHEAD + dk]);
}

// ---------------------------------------------------------------------------
// WMMA GEMM with async-LDS A staging:
//   block = 256 threads (8 waves) -> one 32-row A tile x 512 output columns.
//   Wave 0 double-buffers the 32x64 bf16 A slab into LDS with
//   global_load_async_to_lds_b128 (ASYNCcnt), all waves consume it as
//   WMMA A-fragments via ds loads; B comes from pre-packed global fragments.
// epi: 0 = f32 out, 1 = bf16 out, 2 = relu -> bf16 out,
//      3 = bf16 transposed V out: [B,H,DK,S]
// ---------------------------------------------------------------------------
__device__ __forceinline__ void stage_a_slab(const unsigned short* Arow, int K, int ks,
                                             unsigned short* dst, int lane) {
    // 32 rows x 64 cols bf16 = 4KB = 256 b128 chunks = 8 async instrs (32 lanes).
#pragma unroll
    for (int i = 0; i < 8; ++i) {
        int c   = i * 32 + lane;
        int row = c >> 3;
        int col = (c & 7) << 3;
        async_load_b128((unsigned int)(size_t)(dst + row * 64 + col),
                        Arow + (size_t)row * K + ks + col);
    }
}

__global__ __launch_bounds__(256) void gemm_wmma_kernel(
    const unsigned short* __restrict__ A, const unsigned short* __restrict__ Bp,
    const float* __restrict__ bias, void* __restrict__ out,
    int M, int N, int K, int epi) {
    __shared__ __attribute__((aligned(16))) unsigned short As[2][32 * 64]; // 8 KB

    const int nTm = M >> 5;
    int bm = blockIdx.x % nTm;          // 32-row tile index
    int bn = blockIdx.x / nTm;          // 512-column super-tile index
    int w    = threadIdx.x >> 5;
    int lane = threadIdx.x & 31;

    const int ntiles16 = N >> 4;
    const int nsteps   = K >> 6;        // 64-wide K supersteps (2 wmma k-steps each)
    const unsigned short* Arow = A + (size_t)(bm * 32) * K;
    const int ntbase = (bn * 8 + w) * 4;   // this wave's first 16-col B tile

    v8f acc[2][4];
#pragma unroll
    for (int i = 0; i < 2; ++i)
#pragma unroll
        for (int j = 0; j < 4; ++j) acc[i][j] = zero8();

    if (w == 0) {
        stage_a_slab(Arow, K, 0, &As[0][0], lane);
        wait_async0();
    }
    __syncthreads();

    for (int s = 0; s < nsteps; ++s) {
        int cur = s & 1;
        if (w == 0 && s + 1 < nsteps)
            stage_a_slab(Arow, K, (s + 1) << 6, &As[cur ^ 1][0], lane);

#pragma unroll
        for (int kk = 0; kk < 2; ++kk) {
            int ktg = s * 2 + kk;
            v16bf a0 = load_a_frag(&As[cur][0], 64, 0,  kk * 32, lane);
            v16bf a1 = load_a_frag(&As[cur][0], 64, 16, kk * 32, lane);
#pragma unroll
            for (int j = 0; j < 4; ++j) {
                v16bf b = load_bp_frag(Bp, ntiles16, ktg, ntbase + j, lane);
                acc[0][j] = wmma_bf16(a0, b, acc[0][j]);
                acc[1][j] = wmma_bf16(a1, b, acc[1][j]);
            }
        }
        if (w == 0 && s + 1 < nsteps) wait_async0();
        __syncthreads();
    }

    int col = lane & 15;
    int ro  = (lane >> 4) << 3;   // 0 or 8
#pragma unroll
    for (int i = 0; i < 2; ++i) {
#pragma unroll
        for (int j = 0; j < 4; ++j) {
#pragma unroll
            for (int r = 0; r < 8; ++r) {
                int m = bm * 32 + i * 16 + ro + r;
                int n = (ntbase + j) * 16 + col;
                float val = acc[i][j][r] + bias[n];
                if (epi == 0) {
                    ((float*)out)[(size_t)m * N + n] = val;
                } else if (epi == 1) {
                    ((unsigned short*)out)[(size_t)m * N + n] = f2bf(val);
                } else if (epi == 2) {
                    ((unsigned short*)out)[(size_t)m * N + n] = f2bf(fmaxf(val, 0.0f));
                } else { // epi == 3 : V transposed -> [B, H, DK, S]
                    int b  = m >> 10;          // / SEQ
                    int s2 = m & (SEQ - 1);
                    int h  = n >> 6;           // / DHEAD
                    int dv = n & (DHEAD - 1);
                    ((unsigned short*)out)[((((size_t)b * NHEAD + h) * DHEAD + dv) << 10) + s2]
                        = f2bf(val);
                }
            }
        }
    }
}

// ---------------------------------------------------------------------------
// Attention: one block per (b, h, 16-row s-tile); 4 waves / 128 threads.
// Phase 1: logits(16x1024) = Q Kt / sqrt(DK) + mask*(-1e9)  -> LDS (f32)
// Phase 2: row softmax (wave-parallel, shfl reductions)      -> LDS (bf16)
// Phase 3: ctx(16x64) = P @ V via WMMA (A from LDS, B from vT)
// ---------------------------------------------------------------------------
__global__ __launch_bounds__(128) void attention_kernel(
    const unsigned short* __restrict__ q, const unsigned short* __restrict__ k,
    const unsigned short* __restrict__ vT, const float* __restrict__ mask,
    unsigned short* __restrict__ ctx) {
    __shared__ float logits[16 * SEQ];                               // 64 KB
    __shared__ __attribute__((aligned(16))) unsigned short attnP[16 * SEQ]; // 32 KB

    int bid = blockIdx.x;
    int st = bid & 63;          // SEQ/16 = 64 row tiles
    int h  = (bid >> 6) & (NHEAD - 1);
    int b  = bid >> 10;
    int tid = threadIdx.x;
    int w    = tid >> 5;
    int lane = tid & 31;

    const unsigned short* qb = q  + ((size_t)(b * SEQ + st * 16)) * DMODEL + h * DHEAD;
    const unsigned short* kb = k  + (size_t)b * SEQ * DMODEL + h * DHEAD;
    const unsigned short* vb = vT + ((size_t)(b * NHEAD + h)) * DHEAD * SEQ;

    int col = lane & 15;
    int ro  = (lane >> 4) << 3;

    // ---- Phase 1: logits ----
    v16bf aq0 = load_a_frag(qb, DMODEL, 0, 0,  lane);
    v16bf aq1 = load_a_frag(qb, DMODEL, 0, 32, lane);
    for (int nt = w; nt < 64; nt += 4) {
        v8f c = zero8();
        c = wmma_bf16(aq0, load_bt_frag(kb, DMODEL, nt * 16, 0,  lane), c);
        c = wmma_bf16(aq1, load_bt_frag(kb, DMODEL, nt * 16, 32, lane), c);
#pragma unroll
        for (int r = 0; r < 8; ++r) {
            int srow = st * 16 + ro + r;
            int t    = nt * 16 + col;
            logits[(ro + r) * SEQ + t] =
                c[r] * 0.125f + mask[(size_t)srow * SEQ + t] * (-1e9f);
        }
    }
    __syncthreads();

    // ---- Phase 2: softmax, 4 rows per wave ----
    for (int r = w * 4; r < w * 4 + 4; ++r) {
        const float* Lr = &logits[r * SEQ];
        float mx = -3.0e38f;
        for (int i = lane; i < SEQ; i += 32) mx = fmaxf(mx, Lr[i]);
#pragma unroll
        for (int d = 16; d > 0; d >>= 1) mx = fmaxf(mx, __shfl_xor(mx, d, 32));
        float s = 0.0f;
        for (int i = lane; i < SEQ; i += 32) s += __expf(Lr[i] - mx);
#pragma unroll
        for (int d = 16; d > 0; d >>= 1) s += __shfl_xor(s, d, 32);
        float inv = 1.0f / s;
        for (int i = lane; i < SEQ; i += 32)
            attnP[r * SEQ + i] = f2bf(__expf(Lr[i] - mx) * inv);
    }
    __syncthreads();

    // ---- Phase 3: ctx = P @ V ; wave w owns dv tile [w*16, w*16+16) ----
    v8f c = zero8();
    for (int kt = 0; kt < SEQ / 32; ++kt) {
        v16bf a  = load_a_frag(attnP, SEQ, 0, kt * 32, lane);
        v16bf bb = load_bt_frag(vb, SEQ, w * 16, kt * 32, lane);
        c = wmma_bf16(a, bb, c);
    }
#pragma unroll
    for (int r = 0; r < 8; ++r) {
        size_t o = ((size_t)(b * SEQ + st * 16 + ro + r)) * DMODEL
                 + h * DHEAD + w * 16 + col;
        ctx[o] = f2bf(c[r]);
    }
}

// ---------------------------------------------------------------------------
// Residual + LayerNorm over rows of length 1024. One block (256 thr) per row.
// out_f (f32) always written; out_bf (bf16) optionally written for next GEMM.
// ---------------------------------------------------------------------------
__global__ __launch_bounds__(256) void residual_ln_kernel(
    const float* __restrict__ a, const float* __restrict__ b,
    const float* __restrict__ w, const float* __restrict__ bias,
    float* __restrict__ out_f, unsigned short* __restrict__ out_bf) {
    __shared__ float red[256];
    int row = blockIdx.x;
    int tid = threadIdx.x;
    const float* pa = a + (size_t)row * DMODEL;
    const float* pb = b + (size_t)row * DMODEL;

    float loc[4];
    float s = 0.0f;
#pragma unroll
    for (int i = 0; i < 4; ++i) {
        int c = tid + i * 256;
        loc[i] = pa[c] + pb[c];
        s += loc[i];
    }
    red[tid] = s;
    __syncthreads();
    for (int d = 128; d > 0; d >>= 1) {
        if (tid < d) red[tid] += red[tid + d];
        __syncthreads();
    }
    float mean = red[0] * (1.0f / DMODEL);
    __syncthreads();

    float v = 0.0f;
#pragma unroll
    for (int i = 0; i < 4; ++i) {
        float dd = loc[i] - mean;
        v += dd * dd;
    }
    red[tid] = v;
    __syncthreads();
    for (int d = 128; d > 0; d >>= 1) {
        if (tid < d) red[tid] += red[tid + d];
        __syncthreads();
    }
    float rstd = rsqrtf(red[0] * (1.0f / DMODEL) + 1e-5f);

#pragma unroll
    for (int i = 0; i < 4; ++i) {
        int c = tid + i * 256;
        float y = (loc[i] - mean) * rstd * w[c] + bias[c];
        if (out_f)  out_f[(size_t)row * DMODEL + c] = y;
        if (out_bf) out_bf[(size_t)row * DMODEL + c] = f2bf(y);
    }
}

// ---------------------------------------------------------------------------
// Host orchestration
// ---------------------------------------------------------------------------
extern "C" void kernel_launch(void* const* d_in, const int* in_sizes, int n_in,
                              void* d_out, int out_size, void* d_ws, size_t ws_size,
                              hipStream_t stream) {
    (void)in_sizes; (void)n_in; (void)out_size; (void)ws_size;

    const float* src   = (const float*)d_in[0];
    const float* mask  = (const float*)d_in[1];
    const float* Wq    = (const float*)d_in[2];
    const float* bq    = (const float*)d_in[3];
    const float* Wk    = (const float*)d_in[4];
    const float* bk    = (const float*)d_in[5];
    const float* Wv    = (const float*)d_in[6];
    const float* bv    = (const float*)d_in[7];
    const float* Wo    = (const float*)d_in[8];
    const float* bo    = (const float*)d_in[9];
    const float* ln1_w = (const float*)d_in[10];
    const float* ln1_b = (const float*)d_in[11];
    const float* W1    = (const float*)d_in[12];
    const float* b1    = (const float*)d_in[13];
    const float* W2    = (const float*)d_in[14];
    const float* b2    = (const float*)d_in[15];
    const float* ln2_w = (const float*)d_in[16];
    const float* ln2_b = (const float*)d_in[17];
    float* out = (float*)d_out;

    // ---- workspace carve-up ----
    size_t off = 0;
    char* base = (char*)d_ws;
    auto carve = [&](size_t bytes) -> void* {
        void* p = base + off;
        off += (bytes + 255) & ~(size_t)255;
        return p;
    };
    const size_t MD  = (size_t)MROWS * DMODEL;
    const size_t MF  = (size_t)MROWS * DFF;
    unsigned short* src_bf = (unsigned short*)carve(MD * 2);
    unsigned short* q_bf   = (unsigned short*)carve(MD * 2);
    unsigned short* k_bf   = (unsigned short*)carve(MD * 2);
    unsigned short* vT_bf  = (unsigned short*)carve(MD * 2);
    unsigned short* ctx_bf = (unsigned short*)carve(MD * 2);
    unsigned short* x_bf   = (unsigned short*)carve(MD * 2);
    unsigned short* h_bf   = (unsigned short*)carve(MF * 2);
    float*          attn_f = (float*)carve(MD * 4);
    float*          x_f    = (float*)carve(MD * 4);
    float*          ffn_f  = (float*)carve(MD * 4);
    unsigned short* WqP = (unsigned short*)carve((size_t)DMODEL * DMODEL * 2);
    unsigned short* WkP = (unsigned short*)carve((size_t)DMODEL * DMODEL * 2);
    unsigned short* WvP = (unsigned short*)carve((size_t)DMODEL * DMODEL * 2);
    unsigned short* WoP = (unsigned short*)carve((size_t)DMODEL * DMODEL * 2);
    unsigned short* W1P = (unsigned short*)carve((size_t)DMODEL * DFF * 2);
    unsigned short* W2P = (unsigned short*)carve((size_t)DFF * DMODEL * 2);

    // ---- stage 0: conversions & weight packing ----
    {
        int n = (int)MD;
        f32_to_bf16_kernel<<<n / 256, 256, 0, stream>>>(src, src_bf, n);
    }
    {
        int n = DMODEL * DMODEL;
        pack_b_qkv_kernel<<<n / 256, 256, 0, stream>>>(Wq, WqP);
        pack_b_qkv_kernel<<<n / 256, 256, 0, stream>>>(Wk, WkP);
        pack_b_qkv_kernel<<<n / 256, 256, 0, stream>>>(Wv, WvP);
        pack_b_std_kernel<<<n / 256, 256, 0, stream>>>(Wo, WoP, DMODEL, DMODEL);
    }
    {
        int n = DMODEL * DFF;
        pack_b_std_kernel<<<n / 256, 256, 0, stream>>>(W1, W1P, DMODEL, DFF);
        pack_b_std_kernel<<<n / 256, 256, 0, stream>>>(W2, W2P, DFF, DMODEL);
    }

    auto launch_gemm = [&](const unsigned short* A, const unsigned short* Bp,
                           const float* bias, void* o, int M, int N, int K, int epi) {
        int blocks = (M / 32) * (N / 512);   // 8 waves/block, 512 cols/block
        gemm_wmma_kernel<<<blocks, 256, 0, stream>>>(A, Bp, bias, o, M, N, K, epi);
    };

    // ---- stage 1: QKV projections ----
    launch_gemm(src_bf, WqP, bq, q_bf,  MROWS, DMODEL, DMODEL, /*bf16*/1);
    launch_gemm(src_bf, WkP, bk, k_bf,  MROWS, DMODEL, DMODEL, /*bf16*/1);
    launch_gemm(src_bf, WvP, bv, vT_bf, MROWS, DMODEL, DMODEL, /*vT  */3);

    // ---- stage 2: attention ----
    {
        int blocks = BATCH * NHEAD * (SEQ / 16);   // 8192
        attention_kernel<<<blocks, 128, 0, stream>>>(q_bf, k_bf, vT_bf, mask, ctx_bf);
    }

    // ---- stage 3: output projection + LN1 ----
    launch_gemm(ctx_bf, WoP, bo, attn_f, MROWS, DMODEL, DMODEL, /*f32*/0);
    residual_ln_kernel<<<MROWS, 256, 0, stream>>>(src, attn_f, ln1_w, ln1_b, x_f, x_bf);

    // ---- stage 4: FFN + LN2 ----
    launch_gemm(x_bf, W1P, b1, h_bf,  MROWS, DFF, DMODEL, /*relu bf16*/2);
    launch_gemm(h_bf, W2P, b2, ffn_f, MROWS, DMODEL, DFF,  /*f32*/0);
    residual_ln_kernel<<<MROWS, 256, 0, stream>>>(x_f, ffn_f, ln2_w, ln2_b, out, nullptr);
}
